// IndexNet_69518340653115
// MI455X (gfx1250) — compile-verified, matrix operand order checked by reading, not run on hardware
//
#include <hip/hip_runtime.h>
#include <hip/hip_bf16.h>
#include <math.h>

typedef _Float16 half_t;
typedef __attribute__((ext_vector_type(16))) _Float16 v16h;
typedef __attribute__((ext_vector_type(8)))  _Float16 v8h;
typedef __attribute__((ext_vector_type(8)))  float    v8f;

#define N_SAMP 32768
// fragment-region offsets in halves inside d_ws
#define OFF_W2  0         // 32d x 2kt x 4nt frags (512 halves each)
#define OFF_W3  131072    // 32d x 2kt x 8nt
#define OFF_B3  393216    // 8nt  (K=32 = d dimension)
#define OFF_RW1 397312    // 4kt x 4nt
#define OFF_RW2 405504    // 2kt x 4nt
#define OFF_RW3 409600    // 2kt x 8nt
#define OFF_W1H 417792    // W1 (32x64) row-major f16
#define OFF_B1H 419840    // b1 (32x64) row-major f16
#define WS_HALVES 421888  // ~824 KB of d_ws

// k index inside a K=32 fragment for (lane, half-slot i):
// k = (i&7) + (i>=8 ? 16 : 0) + (lane>=16 ? 8 : 0)  -> two contiguous 8-half runs
__device__ __forceinline__ int frag_k(int lane, int i) {
  return (i & 7) + ((i >> 3) << 4) + ((lane >> 4) << 3);
}

__device__ __forceinline__ v8f wmma16(v16h a, v16h b, v8f c) {
  return __builtin_amdgcn_wmma_f32_16x16x32_f16(false, a, false, b, (short)0, c, false, false);
}

__device__ __forceinline__ v16h load_frag(const half_t* base, int fragIdx, int lane) {
  return *(const v16h*)(base + (size_t)fragIdx*512 + lane*16);
}

__device__ __forceinline__ v16h cat16(v8h lo, v8h hi) {
  return __builtin_shufflevector(lo, hi, 0,1,2,3,4,5,6,7,8,9,10,11,12,13,14,15);
}

__device__ __forceinline__ v8h splat8(half_t v) {
  v8h r = {v,v,v,v,v,v,v,v};
  return r;
}

// per-wave LDS scratch ordering only: DS ops of one wave execute in order
// (ISA 7.3), so a compiler fence suffices -- no s_barrier needed.
__device__ __forceinline__ void lds_fence() { asm volatile("" ::: "memory"); }

// ---- Prep: f32 weights -> f16 (pre-swizzled WMMA B-fragments + W1/b1 rows) ----
__global__ __launch_bounds__(256) void prep_kernel(
    const float* __restrict__ W1, const float* __restrict__ b1,
    const float* __restrict__ W2, const float* __restrict__ W3,
    const float* __restrict__ b3, const float* __restrict__ rW1,
    const float* __restrict__ rW2, const float* __restrict__ rW3,
    half_t* __restrict__ ws)
{
  int idx = blockIdx.x * 256 + threadIdx.x;
  if (idx >= WS_HALVES) return;
  int r    = idx & 511;
  int lane = r >> 4, i = r & 15;
  int kk = frag_k(lane, i);
  int nn = lane & 15;
  float v;
  if (idx < OFF_W3) {                        // W2: (32,64,64)
    int f = idx >> 9;
    int nt = f & 3, kt = (f >> 2) & 1, d = f >> 3;
    v = W2[((size_t)d*64 + kk + kt*32)*64 + nn + nt*16];
  } else if (idx < OFF_B3) {                 // W3: (32,64,128)
    int f = (idx - OFF_W3) >> 9;
    int nt = f & 7, kt = (f >> 3) & 1, d = f >> 4;
    v = W3[((size_t)d*64 + kk + kt*32)*128 + nn + nt*16];
  } else if (idx < OFF_RW1) {                // b3: (32,128), K = d
    int nt = (idx - OFF_B3) >> 9;
    v = b3[kk*128 + nn + nt*16];
  } else if (idx < OFF_RW2) {                // rW1: (128,64)
    int f = (idx - OFF_RW1) >> 9;
    int nt = f & 3, kt = f >> 2;
    v = rW1[(kk + kt*32)*64 + nn + nt*16];
  } else if (idx < OFF_RW3) {                // rW2: (64,64)
    int f = (idx - OFF_RW2) >> 9;
    int nt = f & 3, kt = f >> 2;
    v = rW2[(kk + kt*32)*64 + nn + nt*16];
  } else if (idx < OFF_W1H) {                // rW3: (64,128)
    int f = (idx - OFF_RW3) >> 9;
    int nt = f & 7, kt = f >> 3;
    v = rW3[(kk + kt*32)*128 + nn + nt*16];
  } else if (idx < OFF_B1H) {                // W1 rows, plain f16
    v = W1[idx - OFF_W1H];
  } else {                                   // b1 rows, plain f16
    v = b1[idx - OFF_B1H];
  }
  ws[idx] = (half_t)v;
}

// ---- Main: 8 waves/WG, one 16-row tile per wave, 128 rows per workgroup ----
__global__ __launch_bounds__(256) void indexnet_kernel(
    const float* __restrict__ x, const unsigned char* __restrict__ mask,
    const float* __restrict__ b2,
    const float* __restrict__ rb1, const float* __restrict__ rb2,
    const float* __restrict__ rb3,
    const half_t* __restrict__ fr, float* __restrict__ out)
{
  __shared__ float  xs[128*32];      // 16 KB
  __shared__ half_t ms[128*32];      //  8 KB (mask as 0/1 half, row-major)
  __shared__ half_t sc[8][16*128];   // 32 KB, per-wave relayout scratch

  const int tid  = threadIdx.x;
  const int wave = tid >> 5, lane = tid & 31;
  const int R0   = blockIdx.x * 128;

  for (int idx = tid; idx < 128*32; idx += 256) {
    size_t g = (size_t)(R0 + (idx >> 5))*32 + (idx & 31);
    xs[idx] = x[g];
    ms[idx] = mask[g] ? (half_t)1.0f : (half_t)0.0f;
  }
  __syncthreads();           // real cross-wave dependency (staging)

  const int mrow  = lane & 15;   // A-frag row / C-D col within tile
  const int hl    = lane >> 4;   // C/D: rows j vs j+8; A/B: k +8 select
  const int wbase = wave * 16;
  half_t* scw = sc[wave];
  const v8h* w1p = (const v8h*)(fr + OFF_W1H);   // 8-half chunks, 8 per d-row
  const v8h* b1p = (const v8h*)(fr + OFF_B1H);
  const v8h  z8  = {};

  v8f zero8 = {};
  v8f pooled[8];
  #pragma unroll
  for (int t = 0; t < 8; ++t) pooled[t] = zero8;

  for (int d = 0; d < 32; ++d) {
    // ---- h1 = relu(x[:,d] * W1[d] + b1[d]) -> A-frags via packed f16 math ----
    half_t xh = (half_t)xs[(wbase + mrow)*32 + d];
    v8h x8 = splat8(xh);
    v16h a1[2];
    #pragma unroll
    for (int kt = 0; kt < 2; ++kt) {
      int c0 = d*8 + kt*4 + hl;                 // chunk = (d*64 + kt*32 + hl*8)/8
      v8h lo = __builtin_elementwise_max(__builtin_elementwise_fma(w1p[c0],   x8, b1p[c0]),   z8);
      v8h hi = __builtin_elementwise_max(__builtin_elementwise_fma(w1p[c0+2], x8, b1p[c0+2]), z8);
      a1[kt] = cat16(lo, hi);
    }
    // ---- h2 = A1 @ W2[d] (8 WMMAs) ----
    v8f h2c[4];
    #pragma unroll
    for (int nt = 0; nt < 4; ++nt) {
      v8f c = zero8;
      c = wmma16(a1[0], load_frag(fr + OFF_W2, (d*2+0)*4 + nt, lane), c);
      c = wmma16(a1[1], load_frag(fr + OFF_W2, (d*2+1)*4 + nt, lane), c);
      h2c[nt] = c;
    }
    // ---- relu(+b2), relayout D->A through per-wave LDS (f16, row-major) ----
    #pragma unroll
    for (int nt = 0; nt < 4; ++nt) {
      float b2v = b2[d*64 + mrow + nt*16];
      #pragma unroll
      for (int j = 0; j < 8; ++j) {
        float t = h2c[nt][j] + b2v;
        scw[(hl*8 + j)*64 + mrow + nt*16] = (half_t)(t > 0.f ? t : 0.f);
      }
    }
    lds_fence();
    // ---- A2 frags: 2x ds_load_b128 each; mask applied as per-row splat ----
    v8h m8 = splat8(ms[(wbase + mrow)*32 + d]);
    v16h a2[2];
    #pragma unroll
    for (int kt = 0; kt < 2; ++kt) {
      const half_t* pbase = &scw[mrow*64 + hl*8 + kt*32];
      v8h lo = *(const v8h*)(pbase)      * m8;
      v8h hi = *(const v8h*)(pbase + 16) * m8;
      a2[kt] = cat16(lo, hi);
    }
    // ---- pooled += A2 @ W3[d] (16 WMMAs, 16x128 resident in VGPRs) ----
    #pragma unroll
    for (int ot = 0; ot < 8; ++ot) {
      pooled[ot] = wmma16(a2[0], load_frag(fr + OFF_W3, (d*2+0)*8 + ot, lane), pooled[ot]);
      pooled[ot] = wmma16(a2[1], load_frag(fr + OFF_W3, (d*2+1)*8 + ot, lane), pooled[ot]);
    }
    lds_fence();  // keep next iteration's stores after this iteration's loads
  }

  // ---- pooled += maskf(16x32) @ b3(32x128): masked bias in one K=32 pass ----
  {
    const half_t* mb = &ms[(wbase + mrow)*32 + hl*8];
    v16h am = cat16(*(const v8h*)(mb), *(const v8h*)(mb + 16));
    #pragma unroll
    for (int ot = 0; ot < 8; ++ot)
      pooled[ot] = wmma16(am, load_frag(fr + OFF_B3, ot, lane), pooled[ot]);
  }

  // ---- head: 128 -> 64 -> 64 -> 128 (per-wave LDS relayouts, ds_load_b128) ----
  #pragma unroll
  for (int ot = 0; ot < 8; ++ot)
    #pragma unroll
    for (int j = 0; j < 8; ++j)
      scw[(hl*8 + j)*128 + mrow + ot*16] = (half_t)pooled[ot][j];
  lds_fence();

  v16h p1[4];
  #pragma unroll
  for (int kt = 0; kt < 4; ++kt) {
    const half_t* pb = &scw[mrow*128 + hl*8 + kt*32];
    p1[kt] = cat16(*(const v8h*)(pb), *(const v8h*)(pb + 16));
  }
  v8f r1c[4];
  #pragma unroll
  for (int nt = 0; nt < 4; ++nt) {
    v8f c = zero8;
    #pragma unroll
    for (int kt = 0; kt < 4; ++kt)
      c = wmma16(p1[kt], load_frag(fr + OFF_RW1, kt*4 + nt, lane), c);
    r1c[nt] = c;
  }
  lds_fence();
  #pragma unroll
  for (int nt = 0; nt < 4; ++nt) {
    float bv = rb1[mrow + nt*16];
    #pragma unroll
    for (int j = 0; j < 8; ++j) {
      float v = r1c[nt][j] + bv;
      scw[(hl*8 + j)*64 + mrow + nt*16] = (half_t)(v > 0.f ? v : 0.f);
    }
  }
  lds_fence();

  v16h p2[2];
  #pragma unroll
  for (int kt = 0; kt < 2; ++kt) {
    const half_t* pb = &scw[mrow*64 + hl*8 + kt*32];
    p2[kt] = cat16(*(const v8h*)(pb), *(const v8h*)(pb + 16));
  }
  v8f r2c[4];
  #pragma unroll
  for (int nt = 0; nt < 4; ++nt) {
    v8f c = zero8;
    c = wmma16(p2[0], load_frag(fr + OFF_RW2, 0*4 + nt, lane), c);
    c = wmma16(p2[1], load_frag(fr + OFF_RW2, 1*4 + nt, lane), c);
    r2c[nt] = c;
  }
  lds_fence();
  #pragma unroll
  for (int nt = 0; nt < 4; ++nt) {
    float bv = rb2[mrow + nt*16];
    #pragma unroll
    for (int j = 0; j < 8; ++j) {
      float v = r2c[nt][j] + bv;
      scw[(hl*8 + j)*64 + mrow + nt*16] = (half_t)(v > 0.f ? v : 0.f);
    }
  }
  lds_fence();

  v16h p3[2];
  #pragma unroll
  for (int kt = 0; kt < 2; ++kt) {
    const half_t* pb = &scw[mrow*64 + hl*8 + kt*32];
    p3[kt] = cat16(*(const v8h*)(pb), *(const v8h*)(pb + 16));
  }
  #pragma unroll
  for (int ot = 0; ot < 8; ++ot) {
    v8f c = zero8;
    c = wmma16(p3[0], load_frag(fr + OFF_RW3, 0*8 + ot, lane), c);
    c = wmma16(p3[1], load_frag(fr + OFF_RW3, 1*8 + ot, lane), c);
    int n = mrow + ot*16;
    float bv = rb3[n];
    #pragma unroll
    for (int j = 0; j < 8; ++j) {
      int row = R0 + wbase + hl*8 + j;
      float v = c[j] + bv;
      if (n < 64) {
        out[(size_t)row*64 + n] = v;                               // mu
      } else {
        float sp = fmaxf(v, 0.f) + log1pf(expf(-fabsf(v)));        // softplus
        out[(size_t)N_SAMP*64 + (size_t)row*64 + (n - 64)] = sp;   // sigma
      }
    }
  }
}

extern "C" void kernel_launch(void* const* d_in, const int* in_sizes, int n_in,
                              void* d_out, int out_size, void* d_ws, size_t ws_size,
                              hipStream_t stream) {
  const float* x    = (const float*)d_in[0];
  const unsigned char* mask = (const unsigned char*)d_in[1]; // jax bool = 1 byte
  const float* W1  = (const float*)d_in[2];
  const float* b1  = (const float*)d_in[3];
  const float* W2  = (const float*)d_in[4];
  const float* b2  = (const float*)d_in[5];
  const float* W3  = (const float*)d_in[6];
  const float* b3  = (const float*)d_in[7];
  const float* rW1 = (const float*)d_in[8];
  const float* rb1 = (const float*)d_in[9];
  const float* rW2 = (const float*)d_in[10];
  const float* rb2 = (const float*)d_in[11];
  const float* rW3 = (const float*)d_in[12];
  const float* rb3 = (const float*)d_in[13];
  half_t* ws = (half_t*)d_ws;

  prep_kernel<<<WS_HALVES/256, 256, 0, stream>>>(W1, b1, W2, W3, b3,
                                                 rW1, rW2, rW3, ws);
  indexnet_kernel<<<N_SAMP/128, 256, 0, stream>>>(x, mask, b2,
                                                  rb1, rb2, rb3, ws, (float*)d_out);
}
